// Discriminator_80195629351349
// MI455X (gfx1250) — matrix operand order repaired and checked by reading, not compiled, not run on hardware
//
#include <hip/hip_runtime.h>
#include <hip/hip_bf16.h>
#include <math.h>

typedef __attribute__((ext_vector_type(16))) _Float16 v16h;
typedef __attribute__((ext_vector_type(8)))  float    v8f;

#define BB   4
#define CC   8
#define HH   384
#define WW   384
#define CH   3072      // C*H
#define INW  6144      // 2*CH
#define NCH  18        // CHANS
#define MID  192

// d_ws layout in floats
#define OFF_HEAT 16                    // B*W floats (meta ints live in [0,8))
#define OFF_PQ   1568                  // B*W*32 floats (stride-32 rows, cols 18..31 zeroed)
#define OFF_PA   (OFF_PQ + BB*WW*32)   // B*W*32 floats
// total: OFF_PA + BB*WW*32 = 99,872 floats (~400 KB)

// ---------------------------------------------------------------------------
// Kernel 1: per-batch left/right bounds (faithful argmax semantics) + zero heat
// ---------------------------------------------------------------------------
__global__ void k_meta(const float* __restrict__ amask, float* __restrict__ ws) {
  int b = blockIdx.x;
  int*   meta = (int*)ws;
  float* heat = ws + OFF_HEAT;
  for (int i = threadIdx.x; i < WW; i += blockDim.x) heat[b * WW + i] = 0.f;
  if (threadIdx.x == 0) {
    const float* row = amask + b * WW;
    int ir = 0;
    for (int i = 0; i < MID; i++) { if (row[MID + i] < 1.f) { ir = i; break; } }
    int il = 0;
    for (int i = 0; i < MID; i++) { if (row[MID - 1 - i] < 1.f) { il = i; break; } }
    meta[b * 2 + 0] = il + 1;    // left  (reference quirk: index into reversed half, +1)
    meta[b * 2 + 1] = MID + ir;  // right
  }
}

// ---------------------------------------------------------------------------
// Kernel 2: fused |kspace| magnitude + split first-linear projection.
// 64 consecutive columns per block (coalesced 8B/lane), 4 k-partials per
// column reduced through LDS. Tile-level early exit from the masks.
// Output rows are stride-32 with cols 18..31 = 0 (enables branchless WMMA pad).
// ---------------------------------------------------------------------------
__global__ __launch_bounds__(256) void k_proj(
    const float* __restrict__ acq_k,   // acquired_kspace  [B,C,H,W,2]
    const float* __restrict__ acqg_k,  // acquiring_kspace [B,C,H,W,2]
    const float* __restrict__ qmask,   // acquiring_mask   [B,W]
    const float* __restrict__ W1,      // [18, 6144]
    float* __restrict__ ws) {
  int b     = blockIdx.y;
  int w0    = blockIdx.x * 64;
  int which = blockIdx.z;  // 0 = Pq (acquiring), 1 = Pa (acquired)
  const int* meta = (const int*)ws;

  if (which == 0) {
    bool any = false;
    for (int i = 0; i < 64; i++)
      if (qmask[b * WW + w0 + i] > 0.f) { any = true; break; }
    if (!any) return;  // no acquiring column in this tile
  } else {
    int left = meta[b * 2], right = meta[b * 2 + 1];
    if (w0 >= right || w0 + 64 <= left) return;  // outside [left,right)
  }

  const float* src  = (which == 0) ? acqg_k : acq_k;
  const float* Wrow = W1 + ((which == 0) ? 0 : CH);
  float*       P    = ws + ((which == 0) ? OFF_PQ : OFF_PA);

  int col = w0 + (threadIdx.x & 63);
  int sub = threadIdx.x >> 6;  // 0..3 k-partials per column

  float acc[NCH];
#pragma unroll
  for (int j = 0; j < NCH; j++) acc[j] = 0.f;

  const float* basep = src + ((size_t)b * CC * HH * WW + col) * 2;
  for (int k = sub; k < CH; k += 4) {
    float2 v  = *(const float2*)(basep + (size_t)k * WW * 2);
    float mag = sqrtf(v.x * v.x + v.y * v.y);
#pragma unroll
    for (int j = 0; j < NCH; j++) acc[j] += mag * Wrow[j * INW + k];
  }

  __shared__ float red[4][64][NCH];
#pragma unroll
  for (int j = 0; j < NCH; j++) red[sub][threadIdx.x & 63][j] = acc[j];
  __syncthreads();
  if (sub == 0) {
    float o[32];
#pragma unroll
    for (int j = 0; j < NCH; j++)
      o[j] = red[0][threadIdx.x][j] + red[1][threadIdx.x][j] +
             red[2][threadIdx.x][j] + red[3][threadIdx.x][j];
#pragma unroll
    for (int j = NCH; j < 32; j++) o[j] = 0.f;  // zero pad cols 18..31
    float4* out = (float4*)(P + (size_t)(b * WW + col) * 32);
#pragma unroll
    for (int t = 0; t < 8; t++) out[t] = *(const float4*)&o[4 * t];
  }
}

// ---------------------------------------------------------------------------
// Kernel 3: WMMA MLP over a 16(wi) x 16(wc) pair tile, one wave per block.
// Branchless zero-padded fragments; float4 LDS/global fragment traffic.
// ---------------------------------------------------------------------------
__global__ __launch_bounds__(32) void k_mlp(
    const float* __restrict__ qmask,
    const float* __restrict__ W2, const float* __restrict__ b1,
    const float* __restrict__ b2, const float* __restrict__ W3,
    const float* __restrict__ b3, const float* __restrict__ W4,
    const float* __restrict__ b4, float* __restrict__ ws) {
  int b   = blockIdx.z;
  int wi0 = blockIdx.y * 16;
  int wc0 = blockIdx.x * 16;

  const int* meta = (const int*)ws;
  int left = meta[b * 2], right = meta[b * 2 + 1];
  if (wc0 >= right || wc0 + 16 <= left) return;  // uniform: no live wc here

  int lane = threadIdx.x;
  int hi   = lane >> 4;   // half-wave select
  int mrow = lane & 15;   // row (A) / column (B,C,D) index

  // wave-uniform bitmask of acquiring wi rows in this tile
  float qv = qmask[b * WW + wi0 + mrow];
  unsigned qbits = (unsigned)__ballot(lane < 16 && qv > 0.f);
  if (qbits == 0) return;

  __shared__ float s_pq[16][32];   // Pq tile, cols 18..31 zero (from k_proj)
  __shared__ float s_h[16][32];    // activation bounce, pad cols self-maintained
  __shared__ float s_b1p[32], s_b2p[32], s_b3p[32], s_w4p[32];

  const float* Pq   = ws + OFF_PQ;
  const float* Pa   = ws + OFF_PA;
  float*       heat = ws + OFF_HEAT;

  // flat coalesced float4 preload of the 16x32 Pq tile (rows contiguous)
  {
    const float4* srcq = (const float4*)(Pq + (size_t)(b * WW + wi0) * 32);
    float4*       dstq = (float4*)&s_pq[0][0];
#pragma unroll
    for (int t = 0; t < 4; t++) dstq[lane + 32 * t] = srcq[lane + 32 * t];
  }
  s_b1p[lane] = (lane < NCH) ? b1[lane] : 0.f;
  s_b2p[lane] = (lane < NCH) ? b2[lane] : 0.f;
  s_b3p[lane] = (lane < NCH) ? b3[lane] : 0.f;
  s_w4p[lane] = (lane < NCH) ? W4[lane] : 0.f;
  float bias4 = b4[0];
  __syncthreads();

  // Per-lane A-slot K-sets are two contiguous runs: [hi*8, hi*8+8) and +16.
  // Pa(+b1) per lane, constant across row-groups; pad slots are exactly 0.
  float paq[16];
  {
    const float* parow = Pa + (size_t)(b * WW + wc0 + mrow) * 32 + hi * 8;
    float4 p0 = *(const float4*)(parow + 0);
    float4 p1 = *(const float4*)(parow + 4);
    float4 p2 = *(const float4*)(parow + 16);
    float4 p3 = *(const float4*)(parow + 20);
    float pv[16] = {p0.x, p0.y, p0.z, p0.w, p1.x, p1.y, p1.z, p1.w,
                    p2.x, p2.y, p2.z, p2.w, p3.x, p3.y, p3.z, p3.w};
#pragma unroll
    for (int s = 0; s < 16; s++) {
      int K = (s < 8) ? (hi * 8 + s) : (16 + hi * 8 + (s - 8));
      paq[s] = pv[s] + s_b1p[K];
    }
  }

  // B fragments for W2^T / W3^T: clamped loads + select (no exec branches).
  v16h w2f0 = {}, w2f1 = {}, w3f0 = {}, w3f1 = {};
#pragma unroll
  for (int s = 0; s < 16; s++) {
    int K  = (s < 8) ? (hi * 8 + s) : (16 + hi * 8 + (s - 8));
    int Kc = (K < NCH) ? K : (NCH - 1);
    int n1 = mrow + 16;
    bool v0 = (K < NCH);
    bool v1 = (K < NCH) && (n1 < NCH);
    float a0 = W2[mrow * NCH + Kc];
    float a1 = W2[((n1 < NCH) ? n1 : (NCH - 1)) * NCH + Kc];
    float a2 = W3[mrow * NCH + Kc];
    float a3 = W3[((n1 < NCH) ? n1 : (NCH - 1)) * NCH + Kc];
    w2f0[s] = v0 ? (_Float16)a0 : (_Float16)0.f;
    w2f1[s] = v1 ? (_Float16)a1 : (_Float16)0.f;
    w3f0[s] = v0 ? (_Float16)a2 : (_Float16)0.f;
    w3f1[s] = v1 ? (_Float16)a3 : (_Float16)0.f;
  }
  v8f z = {};

  for (int r = 0; r < 16; r++) {
    if (((qbits >> r) & 1u) == 0) continue;  // scalar-uniform skip

    // A fragment: h1 = relu(Pq[wi] + Pa[wc] + b1); pad slots relu(0+0)=0.
    v16h a = {};
    {
      const float* qrow = &s_pq[r][hi * 8];
      float4 q0 = *(const float4*)(qrow + 0);
      float4 q1 = *(const float4*)(qrow + 4);
      float4 q2 = *(const float4*)(qrow + 16);
      float4 q3 = *(const float4*)(qrow + 20);
      float qvv[16] = {q0.x, q0.y, q0.z, q0.w, q1.x, q1.y, q1.z, q1.w,
                       q2.x, q2.y, q2.z, q2.w, q3.x, q3.y, q3.z, q3.w};
#pragma unroll
      for (int s = 0; s < 16; s++)
        a[s] = (_Float16)fmaxf(paq[s] + qvv[s], 0.f);
    }

    v8f c0 = __builtin_amdgcn_wmma_f32_16x16x32_f16(false, a, false, w2f0, (short)0, z, false, false);
    v8f c1 = __builtin_amdgcn_wmma_f32_16x16x32_f16(false, a, false, w2f1, (short)0, z, false, false);

    // D -> LDS transpose: unconditional; lanes n>=18 of c1 write exact zeros,
    // self-maintaining the pad columns 18..31.
#pragma unroll
    for (int j = 0; j < 8; j++) {
      int m = j + 8 * hi;
      s_h[m][mrow]      = fmaxf(c0[j] + s_b2p[mrow], 0.f);
      s_h[m][16 + mrow] = fmaxf(c1[j] + s_b2p[16 + mrow], 0.f);
    }
    __syncthreads();

    v16h a2 = {};
    {
      const float* hrow = &s_h[mrow][hi * 8];
      float4 q0 = *(const float4*)(hrow + 0);
      float4 q1 = *(const float4*)(hrow + 4);
      float4 q2 = *(const float4*)(hrow + 16);
      float4 q3 = *(const float4*)(hrow + 20);
      float hv[16] = {q0.x, q0.y, q0.z, q0.w, q1.x, q1.y, q1.z, q1.w,
                      q2.x, q2.y, q2.z, q2.w, q3.x, q3.y, q3.z, q3.w};
#pragma unroll
      for (int s = 0; s < 16; s++) a2[s] = (_Float16)hv[s];
    }
    __syncthreads();

    v8f d0 = __builtin_amdgcn_wmma_f32_16x16x32_f16(false, a2, false, w3f0, (short)0, z, false, false);
    v8f d1 = __builtin_amdgcn_wmma_f32_16x16x32_f16(false, a2, false, w3f1, (short)0, z, false, false);

#pragma unroll
    for (int j = 0; j < 8; j++) {
      int m = j + 8 * hi;
      s_h[m][mrow]      = fmaxf(d0[j] + s_b3p[mrow], 0.f);
      s_h[m][16 + mrow] = fmaxf(d1[j] + s_b3p[16 + mrow], 0.f);
    }
    __syncthreads();

    // Final 18-wide dot + sigmoid + cmask; float4 row reads (cols 18/19 = 0).
    float contrib = 0.f;
    if (lane < 16) {
      float acc = bias4;
      const float* hrow = &s_h[lane][0];
#pragma unroll
      for (int t = 0; t < 5; t++) {
        float4 hq = *(const float4*)(hrow + 4 * t);
        const float* wq = &s_w4p[4 * t];
        acc += hq.x * wq[0] + hq.y * wq[1] + hq.z * wq[2] + hq.w * wq[3];
      }
      float sg = 1.f / (1.f + expf(-acc));
      int wc = wc0 + lane;
      if (wc >= left && wc < right) contrib = sg;
    }
#pragma unroll
    for (int off = 16; off >= 1; off >>= 1)
      contrib += __shfl_xor(contrib, off, 32);
    if (lane == 0) atomicAdd(&heat[b * WW + wi0 + r], contrib);
    __syncthreads();
  }
}

// ---------------------------------------------------------------------------
// Kernel 4: out[b,h,w] = acquiring_mask ? heat[b,w] / denom[b] : 0
// ---------------------------------------------------------------------------
__global__ __launch_bounds__(256) void k_out(const float* __restrict__ qmask,
                                             const float* __restrict__ ws,
                                             float* __restrict__ out) {
  int idx = blockIdx.x * 256 + threadIdx.x;
  if (idx >= BB * HH * WW) return;
  int w  = idx % WW;
  int b  = idx / (HH * WW);
  const int* meta = (const int*)ws;
  float denom = (float)(meta[b * 2 + 1] - meta[b * 2]);
  float v = 0.f;
  if (qmask[b * WW + w] > 0.f) v = (ws + OFF_HEAT)[b * WW + w] / denom;
  out[idx] = v;
}

extern "C" void kernel_launch(void* const* d_in, const int* in_sizes, int n_in,
                              void* d_out, int out_size, void* d_ws, size_t ws_size,
                              hipStream_t stream) {
  (void)in_sizes; (void)n_in; (void)out_size; (void)ws_size;
  const float* acq   = (const float*)d_in[0];   // acquired_kspace
  const float* acqg  = (const float*)d_in[1];   // acquiring_kspace
  const float* amask = (const float*)d_in[2];   // acquired_mask
  const float* qmask = (const float*)d_in[3];   // acquiring_mask
  const float* W1 = (const float*)d_in[4];
  const float* b1 = (const float*)d_in[5];
  const float* W2 = (const float*)d_in[6];
  const float* b2 = (const float*)d_in[7];
  const float* W3 = (const float*)d_in[8];
  const float* b3 = (const float*)d_in[9];
  const float* W4 = (const float*)d_in[10];
  const float* b4 = (const float*)d_in[11];
  float* ws  = (float*)d_ws;
  float* out = (float*)d_out;

  k_meta<<<dim3(BB), dim3(128), 0, stream>>>(amask, ws);
  k_proj<<<dim3(WW / 64, BB, 2), dim3(256), 0, stream>>>(acq, acqg, qmask, W1, ws);
  k_mlp<<<dim3(WW / 16, WW / 16, BB), dim3(32), 0, stream>>>(qmask, W2, b1, b2, W3, b3, W4, b4, ws);
  k_out<<<dim3((BB * HH * WW + 255) / 256), dim3(256), 0, stream>>>(qmask, ws, out);
}
